// CswinPool_42614665511011
// MI455X (gfx1250) — matrix operand order repaired and checked by reading, not compiled, not run on hardware
//
#include <hip/hip_runtime.h>

// ---------------------------------------------------------------------------
// CSWin cross-shaped window attention for gfx1250 (MI455X), bf16 WMMA path.
//   DIM=256, HEADS=8, P=4, b=16, X=Y=64, head_dim(per half)=16
//   SCALE = (256/8)^-0.5 = 1/sqrt(32)
// ---------------------------------------------------------------------------

#define SCALE_F 0.17677669529663687f
#define BSX 1048576            // 256*4096 elements: batch stride of (b,256,64,64)

typedef __attribute__((ext_vector_type(16))) __bf16          v16bf;
typedef __attribute__((ext_vector_type(16))) unsigned short  v16s;
typedef __attribute__((ext_vector_type(8)))  unsigned short  v8s;
typedef __attribute__((ext_vector_type(8)))  float           v8f;
typedef __attribute__((ext_vector_type(4)))  float           v4f;

__device__ __forceinline__ unsigned short f2bf(float f) {
  unsigned int u = __builtin_bit_cast(unsigned int, f);
  u += 0x7FFFu + ((u >> 16) & 1u);            // round-to-nearest-even
  return (unsigned short)(u >> 16);
}
__device__ __forceinline__ float bf2f(unsigned short h) {
  return __builtin_bit_cast(float, (unsigned int)h << 16);
}
__device__ __forceinline__ v8f v8f_zero() {
  v8f z;
#pragma unroll
  for (int i = 0; i < 8; ++i) z[i] = 0.0f;
  return z;
}

// ---------------------------------------------------------------------------
// Channel GEMM:  Y[b,o,n] = sum_k W[o*256+k] * Xeff[b,k,n]   (M=256, K=256)
//   MODE 0: Xeff[k,n] = X[b,k,pix=n]                (1x1 conv)
//   MODE 1: qv_h  k=(c,dx): X[b,c,(2*xo+dx)*64+y],  n=xo*64+y   (out 32x64)
//   MODE 2: qv_v  k=(c,dy): X[b,128+c,x*64+2*yo+dy],n=x*32+yo   (out 64x32)
// PROJ: f32 output + residual; else bf16 output.
// Block: 256 threads (8 waves); tile M=128 x N=64; full K staged in LDS.
// ---------------------------------------------------------------------------
template <int MODE, bool XBF16, bool PROJ>
__global__ __launch_bounds__(256) void gemm_wmma_kernel(
    const float* __restrict__ W,
    const void*  __restrict__ Xv,
    unsigned short* __restrict__ Yb,
    float* __restrict__ Yf, const float* __restrict__ resid,
    int Nper)
{
  __shared__ alignas(32) unsigned short ldsX[64][272];   // [n][k], 32B-aligned rows

  const int t    = threadIdx.x;
  const int lane = t & 31;
  const int w    = t >> 5;
  const int hi   = lane >> 4;
  const int lo   = lane & 15;
  const int n0   = blockIdx.x * 64;
  const int m0   = blockIdx.y * 128;
  const int b    = blockIdx.z;

  // ---- stage X tile (64 n x 256 k), transposed, converted to bf16 ----
  if (MODE == 0 && XBF16) {
    const unsigned short* Xb = (const unsigned short*)Xv;
    for (int it = 0; it < 8; ++it) {             // 8 bf16 per lane per iter
      int e = (it * 256 + t) * 8;
      int k = e >> 6, n = e & 63;
      const v8s v = *(const v8s*)&Xb[b * BSX + k * 4096 + n0 + n];
#pragma unroll
      for (int i = 0; i < 8; ++i) ldsX[n + i][k] = v[i];
    }
  } else if (MODE == 0 || MODE == 1) {
    const float* Xf = (const float*)Xv;
    for (int it = 0; it < 16; ++it) {            // float4 per lane per iter
      int e = (it * 256 + t) * 4;
      int k = e >> 6, n = e & 63;
      int ga;
      if (MODE == 0) {
        ga = b * BSX + k * 4096 + n0 + n;
      } else {                                   // qv_h: tile = one xo row
        int xo = n0 >> 6;
        int c = k >> 1, dx = k & 1;
        ga = b * BSX + c * 4096 + (2 * xo + dx) * 64 + n;
      }
      const v4f v = *(const v4f*)&Xf[ga];
#pragma unroll
      for (int i = 0; i < 4; ++i) ldsX[n + i][k] = f2bf(v[i]);
    }
  } else {                                       // qv_v: stride-2 gather
    const float* Xf = (const float*)Xv;
    for (int it = 0; it < 64; ++it) {
      int idx = it * 256 + t;
      int k = idx >> 6, n = idx & 63;
      int nG = n0 + n;
      int x = nG >> 5, yo = nG & 31;
      int c = k >> 1, dy = k & 1;
      ldsX[n][k] = f2bf(Xf[b * BSX + (128 + c) * 4096 + x * 64 + 2 * yo + dy]);
    }
  }
  __syncthreads();

  v8f acc[4];
#pragma unroll
  for (int nt = 0; nt < 4; ++nt) acc[nt] = v8f_zero();

  const int om = m0 + w * 16 + lo;              // A-matrix row (output channel)
  const float* wrow = W + om * 256;

  for (int kc = 0; kc < 256; kc += 32) {
    if (kc + 64 <= 256) __builtin_prefetch(wrow + kc + 32, 0, 0);
    // A fragment (16x32 bf16): lanes 0-15 hold K{0..15}, 16-31 K{16..31}
    v16s a;
    const int kw = kc + hi * 16;
#pragma unroll
    for (int i = 0; i < 8; ++i) {
      a[i]     = f2bf(wrow[kw + i]);
      a[i + 8] = f2bf(wrow[kw + 8 + i]);
    }
    const v16bf af = __builtin_bit_cast(v16bf, a);
#pragma unroll
    for (int nt = 0; nt < 4; ++nt) {
      const int n = nt * 16 + lo;
      const v16s  bs = *(const v16s*)&ldsX[n][kc + hi * 16];
      const v16bf bf = __builtin_bit_cast(v16bf, bs);
      acc[nt] = __builtin_amdgcn_wmma_f32_16x16x32_bf16(
          false, af, false, bf, (short)0, acc[nt], false, false);
    }
  }

  // ---- store C tiles (VGPR j: rows j / j+8 for the two lane halves) ----
#pragma unroll
  for (int nt = 0; nt < 4; ++nt) {
#pragma unroll
    for (int j = 0; j < 8; ++j) {
      const int o  = m0 + w * 16 + j + hi * 8;
      const int n  = n0 + nt * 16 + lo;
      const int oi = (b * 256 + o) * Nper + n;
      const float v = acc[nt][j];
      if (PROJ) Yf[oi] = v + resid[oi];
      else      Yb[oi] = f2bf(v);
    }
  }
}

// ---------------------------------------------------------------------------
// Window/unwindow address helpers (bf16 intermediates, element indices).
//   branch 0 (horizontal): window = (b, gy);  branch 1 (vertical): (b, gx)
// ---------------------------------------------------------------------------
__device__ __forceinline__ int qv_addr(int br, int b, int win, int ch, int q) {
  if (br == 0) { int i1 = q >> 2, i2 = q & 3;
                 return (b * 256 + ch) * 2048 + i1 * 64 + win * 4 + i2; }
  else         { int i1 = q >> 5, i2 = q & 31;
                 return (b * 256 + ch) * 2048 + (win * 4 + i1) * 32 + i2; }
}
__device__ __forceinline__ int k_addr(int br, int b, int win, int ch, int n) {
  if (br == 0) { int i1 = n >> 2, i2 = n & 3;
                 return (b * 256 + ch) * 4096 + i1 * 64 + win * 4 + i2; }
  else         { int i1 = n >> 6, i2 = n & 63;
                 return (b * 256 + ch) * 4096 + (win * 4 + i1) * 64 + i2; }
}

// ---------------------------------------------------------------------------
// Attention: one block per (window, head, branch). Nq=128, Nk=256, d=16.
//   S = softmax_k( (Q*SCALE) K^T ) ;  O = S^T V + lepe  (indexed by key tokens)
// d=16 padded to K=32 for WMMA; the padding is stored as zeros in LDS so
// every fragment load is a plain, unpredicated vector load (no EXEC masking
// in the WMMA loops).
// 8 waves: each owns 16 q-rows for S+softmax, then 2 of 16 O tiles.
// ---------------------------------------------------------------------------
__global__ __launch_bounds__(256) void attn_wmma_kernel(
    const unsigned short* __restrict__ Kbuf,
    const unsigned short* __restrict__ Lbuf,
    const unsigned short* __restrict__ QVh,
    const unsigned short* __restrict__ QVv,
    unsigned short* __restrict__ Cat)
{
  __shared__ alignas(32) unsigned short ldsQ[128][32];   // [q][d0..15 | zeros]
  __shared__ alignas(32) unsigned short ldsK[256][32];   // [n][d0..15 | zeros]
  __shared__ alignas(32) unsigned short ldsVT[16][144];  // [d][q]  (transposed)
  __shared__ alignas(32) unsigned short ldsS[256][136];  // [n][q]  softmax(S)^T

  const int t    = threadIdx.x;
  const int lane = t & 31;
  const int w    = t >> 5;
  const int hi   = lane >> 4;
  const int lo   = lane & 15;
  const int head = blockIdx.y;
  const int br   = blockIdx.z;
  const int b    = blockIdx.x >> 4;
  const int win  = blockIdx.x & 15;

  const unsigned short* QV = br ? QVv : QVh;
  const int chQ = head * 16;                    // qv first half
  const int chV = 128 + head * 16;              // qv second half
  const int chK = (br ? 128 : 0) + head * 16;   // k/lepe/cat half

  // ---- stage Q (x SCALE), V^T, K into LDS as bf16; zero the d-padding ----
  for (int it = 0; it < 8; ++it) {
    int e = it * 256 + t;                       // 128 q x 16 d
    int q = e >> 4, d = e & 15;
    ldsQ[q][d]      = f2bf(bf2f(QV[qv_addr(br, b, win, chQ + d, q)]) * SCALE_F);
    ldsQ[q][16 + d] = 0;
  }
  for (int it = 0; it < 8; ++it) {
    int e = it * 256 + t;
    int q = e & 127, d = e >> 7;
    ldsVT[d][q] = QV[qv_addr(br, b, win, chV + d, q)];
  }
  for (int it = 0; it < 16; ++it) {
    int e = it * 256 + t;                       // 256 n x 16 d
    int n = e >> 4, d = e & 15;
    ldsK[n][d]      = Kbuf[k_addr(br, b, win, chK + d, n)];
    ldsK[n][16 + d] = 0;
  }
  __syncthreads();

  // ---- phase 2: S tiles + softmax for this wave's 16 q-rows ----
  const int qbase = w * 16;
  v16s qa;
  {
    const v8s q0 = *(const v8s*)&ldsQ[qbase + lo][hi * 16];       // K{0..7}/{16..23}
    const v8s q1 = *(const v8s*)&ldsQ[qbase + lo][hi * 16 + 8];   // K{8..15}/{24..31}
#pragma unroll
    for (int i = 0; i < 8; ++i) { qa[i] = q0[i]; qa[i + 8] = q1[i]; }
  }
  const v16bf qaf = __builtin_bit_cast(v16bf, qa);

  v8f s[16];
#pragma unroll
  for (int nk = 0; nk < 16; ++nk) {
    // lanes 0-15: d=0..15 of token nk*16+lo ; lanes 16-31: stored zero padding
    const v16s  kb  = *(const v16s*)&ldsK[nk * 16 + lo][hi * 16];
    const v16bf kbf = __builtin_bit_cast(v16bf, kb);
    s[nk] = __builtin_amdgcn_wmma_f32_16x16x32_bf16(
        false, qaf, false, kbf, (short)0, v8f_zero(), false, false);
  }

  {
    float mx[8], sm[8];
#pragma unroll
    for (int j = 0; j < 8; ++j) mx[j] = -3.0e38f;
    for (int nk = 0; nk < 16; ++nk)
#pragma unroll
      for (int j = 0; j < 8; ++j) mx[j] = fmaxf(mx[j], s[nk][j]);
#pragma unroll
    for (int j = 0; j < 8; ++j) {               // reduce within 16-lane halves
      float m = mx[j];
      m = fmaxf(m, __shfl_xor(m, 1, 32));
      m = fmaxf(m, __shfl_xor(m, 2, 32));
      m = fmaxf(m, __shfl_xor(m, 4, 32));
      m = fmaxf(m, __shfl_xor(m, 8, 32));
      mx[j] = m;
      sm[j] = 0.0f;
    }
    for (int nk = 0; nk < 16; ++nk)
#pragma unroll
      for (int j = 0; j < 8; ++j) {
        float e = __expf(s[nk][j] - mx[j]);
        s[nk][j] = e;
        sm[j] += e;
      }
#pragma unroll
    for (int j = 0; j < 8; ++j) {
      float m = sm[j];
      m += __shfl_xor(m, 1, 32);
      m += __shfl_xor(m, 2, 32);
      m += __shfl_xor(m, 4, 32);
      m += __shfl_xor(m, 8, 32);
      sm[j] = 1.0f / m;
    }
    for (int nk = 0; nk < 16; ++nk)
#pragma unroll
      for (int j = 0; j < 8; ++j) {
        const int q = qbase + j + hi * 8;       // C-tile row for this lane half
        const int n = nk * 16 + lo;
        ldsS[n][q] = f2bf(s[nk][j] * sm[j]);
      }
  }
  __syncthreads();

  // ---- phase 3: O = S^T V + lepe ; wave computes 2 of 16 (nk x d) tiles ----
#pragma unroll
  for (int tt = 0; tt < 2; ++tt) {
    const int nkb = (w * 2 + tt) * 16;
    v8f acc;
#pragma unroll
    for (int j = 0; j < 8; ++j) {
      const int tok = nkb + j + hi * 8;
      acc[j] = bf2f(Lbuf[k_addr(br, b, win, chK + lo, tok)]);
    }
#pragma unroll
    for (int qc = 0; qc < 4; ++qc) {            // K=32 chunks over q=128
      v16s a;
      {
        const v8s s0 = *(const v8s*)&ldsS[nkb + lo][qc * 32 + hi * 16];
        const v8s s1 = *(const v8s*)&ldsS[nkb + lo][qc * 32 + hi * 16 + 8];
#pragma unroll
        for (int i = 0; i < 8; ++i) { a[i] = s0[i]; a[i + 8] = s1[i]; }
      }
      const v16bf af = __builtin_bit_cast(v16bf, a);
      const v16s  vs = *(const v16s*)&ldsVT[lo][qc * 32 + hi * 16];
      const v16bf vf = __builtin_bit_cast(v16bf, vs);
      acc = __builtin_amdgcn_wmma_f32_16x16x32_bf16(
          false, af, false, vf, (short)0, acc, false, false);
    }
#pragma unroll
    for (int j = 0; j < 8; ++j) {
      const int tok = nkb + j + hi * 8;
      Cat[k_addr(br, b, win, chK + lo, tok)] = f2bf(acc[j]);
    }
  }
}

// ---------------------------------------------------------------------------
// Launch: 4 projection GEMMs -> attention (4096 window-heads) -> proj+residual
// Workspace (bf16 intermediates, element counts):
//   k_all 16*256*4096, lepe same, qv_h/qv_v 16*256*2048 each, cat 16*256*4096
// ---------------------------------------------------------------------------
extern "C" void kernel_launch(void* const* d_in, const int* in_sizes, int n_in,
                              void* d_out, int out_size, void* d_ws, size_t ws_size,
                              hipStream_t stream) {
  (void)in_sizes; (void)n_in; (void)out_size; (void)ws_size;
  const float* fmap   = (const float*)d_in[0];
  const float* W_qvh  = (const float*)d_in[1];
  const float* W_qvv  = (const float*)d_in[2];
  const float* W_k    = (const float*)d_in[3];
  const float* W_lepe = (const float*)d_in[4];
  const float* W_proj = (const float*)d_in[5];
  float* out = (float*)d_out;

  char* ws = (char*)d_ws;
  unsigned short* ws_k    = (unsigned short*)(ws);                 // 33,554,432 B
  unsigned short* ws_lepe = (unsigned short*)(ws + 33554432);      // 33,554,432 B
  unsigned short* ws_qvh  = (unsigned short*)(ws + 67108864);      // 16,777,216 B
  unsigned short* ws_qvv  = (unsigned short*)(ws + 83886080);      // 16,777,216 B
  unsigned short* ws_cat  = (unsigned short*)(ws + 100663296);     // 33,554,432 B

  const dim3 blk(256);
  gemm_wmma_kernel<0, false, false><<<dim3(64, 2, 16), blk, 0, stream>>>(
      W_k, fmap, ws_k, nullptr, nullptr, 4096);
  gemm_wmma_kernel<0, false, false><<<dim3(64, 2, 16), blk, 0, stream>>>(
      W_lepe, fmap, ws_lepe, nullptr, nullptr, 4096);
  gemm_wmma_kernel<1, false, false><<<dim3(32, 2, 16), blk, 0, stream>>>(
      W_qvh, fmap, ws_qvh, nullptr, nullptr, 2048);
  gemm_wmma_kernel<2, false, false><<<dim3(32, 2, 16), blk, 0, stream>>>(
      W_qvv, fmap, ws_qvv, nullptr, nullptr, 2048);

  attn_wmma_kernel<<<dim3(256, 8, 2), blk, 0, stream>>>(
      ws_k, ws_lepe, ws_qvh, ws_qvv, ws_cat);

  gemm_wmma_kernel<0, true, true><<<dim3(64, 2, 16), blk, 0, stream>>>(
      W_proj, ws_cat, nullptr, out, fmap, 4096);
}
